// NeuronQwen25OmniAudioEncoder_24017457119789
// MI455X (gfx1250) — matrix operand order repaired and checked by reading, not compiled, not run on hardware
//
#include <hip/hip_runtime.h>
#include <hip/hip_bf16.h>
#include <math.h>

typedef __bf16 bf16;
typedef __bf16 v16bf __attribute__((ext_vector_type(16)));
typedef float  v8f   __attribute__((ext_vector_type(8)));

#define WMMA_BF16(a, b, c) \
  __builtin_amdgcn_wmma_f32_16x16x32_bf16(false, (a), false, (b), (short)0, (c), false, false)

struct Frag { union { uint4 u[2]; v16bf v; }; };

__device__ __forceinline__ float geluf(float x) {
  return 0.5f * x * (1.0f + erff(x * 0.70710678118654752f));
}

// gfx1250 async global->LDS copy (ASYNCcnt-tracked, no VGPR staging).
__device__ __forceinline__ void async_load_b128(unsigned int lds_addr, const void* gptr) {
  asm volatile("global_load_async_to_lds_b128 %0, %1, off"
               :: "v"(lds_addr), "v"((unsigned long long)(uintptr_t)gptr)
               : "memory");
}
__device__ __forceinline__ void wait_async_le4() {
  asm volatile("s_wait_asynccnt 0x4" ::: "memory");
}
__device__ __forceinline__ void wait_async_0() {
  asm volatile("s_wait_asynccnt 0x0" ::: "memory");
}
__device__ __forceinline__ unsigned int lds_addr_of(const void* p) {
  return (unsigned int)(uintptr_t)p;   // LDS aperture: addr[31:0] = LDS offset
}

// ---------------------------------------------------------------------------
// bf16 GEMM: out[M,N] = epilogue(A[M,K] @ W[N,K]^T + bias) (+ res)
// act: 0 = none, 1 = exact GELU. Block = 256 threads (8 waves), tile 128x128,
// each wave owns a 32x64 sub-tile (8 WMMAs / K-step). Double-buffered LDS
// filled with global_load_async_to_lds_b128. Requires N % 128 == 0, K % 32 == 0.
// Out-of-range M rows are clamped on load (their C rows are never stored).
// ---------------------------------------------------------------------------
__global__ __launch_bounds__(256)
void gemm_bf16_kernel(const bf16* __restrict__ A, const bf16* __restrict__ W,
                      const bf16* __restrict__ bias, const bf16* __restrict__ res,
                      bf16* __restrict__ out, int M, int N, int K, int act)
{
  __shared__ bf16 As[2][128][32];
  __shared__ bf16 Ws[2][128][32];

  const int tid  = threadIdx.x;
  const int lane = tid & 31;
  const int wave = tid >> 5;
  const int bm   = blockIdx.x * 128;
  const int bn   = blockIdx.y * 128;
  const int wm   = (wave >> 1) * 32;      // wave M offset in tile (0,32,64,96)
  const int wn   = (wave & 1) * 64;       // wave N offset in tile (0,64)
  const int lr   = lane & 15;
  const int kh   = (lane >> 4) * 8;       // A-fragment K base (ISA layout)
  const int kb   = (lane >> 4) * 16;      // unused (B fragments load with kh)

  // per-thread tile-load coordinates: 512 uint4 per matrix, 2 per thread
  const int r0 = (tid)       >> 2, c0 = ((tid)       & 3) * 8;
  const int r1 = (tid + 256) >> 2, c1 = ((tid + 256) & 3) * 8;
  int gmr0 = bm + r0; if (gmr0 > M - 1) gmr0 = M - 1;   // clamp (no zero-fill)
  int gmr1 = bm + r1; if (gmr1 > M - 1) gmr1 = M - 1;
  const int gnr0 = bn + r0;               // N % 128 == 0: always in range
  const int gnr1 = bn + r1;
  const bf16* Ar0 = A + (size_t)gmr0 * K;
  const bf16* Ar1 = A + (size_t)gmr1 * K;
  const bf16* Wr0 = W + (size_t)gnr0 * K;
  const bf16* Wr1 = W + (size_t)gnr1 * K;

  auto issue_tile = [&](int buf, int kt) {
    async_load_b128(lds_addr_of(&As[buf][r0][c0]), Ar0 + kt + c0);
    async_load_b128(lds_addr_of(&As[buf][r1][c1]), Ar1 + kt + c1);
    async_load_b128(lds_addr_of(&Ws[buf][r0][c0]), Wr0 + kt + c0);
    async_load_b128(lds_addr_of(&Ws[buf][r1][c1]), Wr1 + kt + c1);
  };

  v8f acc[2][4];
  #pragma unroll
  for (int i = 0; i < 2; ++i)
    #pragma unroll
    for (int j = 0; j < 4; ++j) acc[i][j] = (v8f){};

  issue_tile(0, 0);                        // prologue

  const int nsteps = K >> 5;
  for (int s = 0; s < nsteps; ++s) {
    const int cur = s & 1;
    const int kt  = s << 5;
    if (s + 1 < nsteps) {
      issue_tile(cur ^ 1, kt + 32);        // fill next buffer
      wait_async_le4();                    // drain this step's tile (in-order)
      if (kt + 64 < K) {                   // near-scope prefetch one more ahead
        __builtin_prefetch(Ar0 + kt + 64 + c0, 0, 3);
        __builtin_prefetch(Wr0 + kt + 64 + c0, 0, 3);
      }
    } else {
      wait_async_0();
    }
    __syncthreads();

    Frag a0, a1, b[4];
    a0.u[0] = *(const uint4*)&As[cur][wm      + lr][kh];
    a0.u[1] = *(const uint4*)&As[cur][wm      + lr][kh + 16];
    a1.u[0] = *(const uint4*)&As[cur][wm + 16 + lr][kh];
    a1.u[1] = *(const uint4*)&As[cur][wm + 16 + lr][kh + 16];
    #pragma unroll
    for (int j = 0; j < 4; ++j) {
      b[j].u[0] = *(const uint4*)&Ws[cur][wn + 16 * j + lr][kh];
      b[j].u[1] = *(const uint4*)&Ws[cur][wn + 16 * j + lr][kh + 16];
    }
    #pragma unroll
    for (int j = 0; j < 4; ++j) {
      acc[0][j] = WMMA_BF16(a0.v, b[j].v, acc[0][j]);
      acc[1][j] = WMMA_BF16(a1.v, b[j].v, acc[1][j]);
    }
    __syncthreads();                       // protect buf reuse next iteration
  }

  const int hi = lane >> 4;
  auto store_tile = [&](v8f c, int m0, int n0) {
    int n = n0 + lr;
    float bvv = bias ? (float)bias[n] : 0.0f;
    #pragma unroll
    for (int r = 0; r < 8; ++r) {
      int m = m0 + r + 8 * hi;
      if (m < M) {
        float v = c[r] + bvv;
        if (act == 1) v = geluf(v);
        if (res) v += (float)res[(size_t)m * N + n];
        out[(size_t)m * N + n] = (bf16)v;
      }
    }
  };
  #pragma unroll
  for (int i = 0; i < 2; ++i)
    #pragma unroll
    for (int j = 0; j < 4; ++j)
      store_tile(acc[i][j], bm + wm + 16 * i, bn + wn + 16 * j);
}

// ---------------------------------------------------------------------------
// Per-(chunk, head) attention: S=100 tokens, head dim 64, block-diagonal.
// scores = Q K^T * 0.125 -> softmax(f32) -> P V, all matmuls via WMMA.
// Block = 128 threads (4 waves); 7 query strips of 16 rows (pad 100->112).
// ---------------------------------------------------------------------------
__global__ __launch_bounds__(128)
void attn_kernel(const bf16* __restrict__ q, const bf16* __restrict__ k,
                 const bf16* __restrict__ v, bf16* __restrict__ o)
{
  const int S = 100, HD = 64, Dm = 1280;
  __shared__ bf16  Vt[64][128];         // V transposed: Vt[n][kk], zero-padded
  __shared__ float Ss[4][16][112];      // per-wave raw scores
  __shared__ bf16  Sb[4][16][128];      // per-wave softmaxed probs (padded)

  const int chunk = blockIdx.x;
  const int head  = blockIdx.y;
  const int tid   = threadIdx.x;
  const int lane  = tid & 31;
  const int wave  = tid >> 5;
  const int lr    = lane & 15;
  const int kh    = (lane >> 4) * 8;    // A-fragment K base
  const int kb    = (lane >> 4) * 16;   // unused (B fragments load with kh)
  const size_t base = (size_t)chunk * S * Dm + (size_t)head * HD;

  for (int i = tid; i < 64 * 128; i += 128) {
    int n  = i >> 7;
    int kk = i & 127;
    Vt[n][kk] = (kk < S) ? v[base + (size_t)kk * Dm + n] : (bf16)0.0f;
  }
  __syncthreads();

  for (int it = 0; it < 2; ++it) {
    int strip = it * 4 + wave;
    bool active = (strip < 7);

    if (active) {                         // --- scores: Q K^T ---
      int m0 = strip * 16;
      int qrow = m0 + lr; if (qrow > S - 1) qrow = S - 1;
      Frag a0, a1;
      a0.u[0] = *(const uint4*)&q[base + (size_t)qrow * Dm + 0  + kh];
      a0.u[1] = *(const uint4*)&q[base + (size_t)qrow * Dm + 16 + kh];
      a1.u[0] = *(const uint4*)&q[base + (size_t)qrow * Dm + 32 + kh];
      a1.u[1] = *(const uint4*)&q[base + (size_t)qrow * Dm + 48 + kh];
      for (int nt = 0; nt < 7; ++nt) {
        int krow = nt * 16 + lr; if (krow > S - 1) krow = S - 1;
        Frag b0, b1;                      // B of K^T = contiguous rows of K
        b0.u[0] = *(const uint4*)&k[base + (size_t)krow * Dm + 0  + kh];
        b0.u[1] = *(const uint4*)&k[base + (size_t)krow * Dm + 16 + kh];
        b1.u[0] = *(const uint4*)&k[base + (size_t)krow * Dm + 32 + kh];
        b1.u[1] = *(const uint4*)&k[base + (size_t)krow * Dm + 48 + kh];
        v8f c = {};
        c = WMMA_BF16(a0.v, b0.v, c);
        c = WMMA_BF16(a1.v, b1.v, c);
        int hi = lane >> 4;
        #pragma unroll
        for (int r = 0; r < 8; ++r)
          Ss[wave][r + 8 * hi][nt * 16 + lr] = c[r] * 0.125f;  // HD^-0.5
      }
    }
    __syncthreads();

    if (active && lane < 16) {            // --- softmax (fp32) ---
      int m = lr;
      float mx = -1e30f;
      for (int c2 = 0; c2 < S; ++c2) mx = fmaxf(mx, Ss[wave][m][c2]);
      float sum = 0.0f;
      for (int c2 = 0; c2 < S; ++c2) {
        float e = __expf(Ss[wave][m][c2] - mx);
        Ss[wave][m][c2] = e;
        sum += e;
      }
      float inv = 1.0f / sum;
      for (int c2 = 0; c2 < S; ++c2)   Sb[wave][m][c2] = (bf16)(Ss[wave][m][c2] * inv);
      for (int c2 = S; c2 < 128; ++c2) Sb[wave][m][c2] = (bf16)0.0f;
    }
    __syncthreads();

    if (active) {                         // --- output: P V ---
      int m0 = strip * 16;
      int hi = lane >> 4;
      #pragma unroll
      for (int nt = 0; nt < 4; ++nt) {
        v8f c = {};
        #pragma unroll
        for (int k0 = 0; k0 < 128; k0 += 32) {
          Frag a, b;
          a.u[0] = *(const uint4*)&Sb[wave][lr][k0 + kh];
          a.u[1] = *(const uint4*)&Sb[wave][lr][k0 + kh + 16];
          b.u[0] = *(const uint4*)&Vt[nt * 16 + lr][k0 + kh];
          b.u[1] = *(const uint4*)&Vt[nt * 16 + lr][k0 + kh + 16];
          c = WMMA_BF16(a.v, b.v, c);
        }
        #pragma unroll
        for (int r = 0; r < 8; ++r) {
          int m = m0 + r + 8 * hi;
          if (m < S) o[base + (size_t)m * Dm + nt * 16 + lr] = (bf16)c[r];
        }
      }
    }
    __syncthreads();
  }
}

// ---------------------------------------------------------------------------
// fp32 LayerNorm over rows of length Dd. One block (128 threads) per row.
// ---------------------------------------------------------------------------
__global__ __launch_bounds__(128)
void ln_kernel(const bf16* __restrict__ x, bf16* __restrict__ y,
               const float* __restrict__ g, const float* __restrict__ b, int Dd)
{
  const int row = blockIdx.x;
  const int tid = threadIdx.x;
  const bf16* xr = x + (size_t)row * Dd;
  float s = 0.0f, ss = 0.0f;
  for (int i = tid; i < Dd; i += 128) {
    float v = (float)xr[i];
    s += v; ss += v * v;
  }
  __shared__ float r1[128], r2[128];
  r1[tid] = s; r2[tid] = ss;
  __syncthreads();
  for (int st = 64; st > 0; st >>= 1) {
    if (tid < st) { r1[tid] += r1[tid + st]; r2[tid] += r2[tid + st]; }
    __syncthreads();
  }
  float mean = r1[0] / (float)Dd;
  float var  = r2[0] / (float)Dd - mean * mean;
  float inv  = rsqrtf(var + 1e-5f);
  bf16* yr = y + (size_t)row * Dd;
  for (int i = tid; i < Dd; i += 128)
    yr[i] = (bf16)(((float)xr[i] - mean) * inv * g[i] + b[i]);
}

// ---------------------------------------------------------------------------
// im2col for conv1: mel (128, 24000) -> im (24000, 384), pad 1, k=3.
// ---------------------------------------------------------------------------
__global__ void pack_conv1_kernel(const bf16* __restrict__ mel, bf16* __restrict__ im)
{
  int i = blockIdx.x * blockDim.x + threadIdx.x;
  if (i >= 24000 * 384) return;
  int row = i / 384, col = i - row * 384;
  int c = row / 200, t = row - c * 200;
  int ic = col / 3, kt = col - ic * 3;
  int u = t + kt - 1;
  bf16 val = (bf16)0.0f;
  if (u >= 0 && u < 200) val = mel[(size_t)ic * 24000 + (size_t)c * 200 + u];
  im[i] = val;
}

// ---------------------------------------------------------------------------
// im2col for conv2 (stride 2): x1 (24000, 1280) token-major -> im (12000, 3840)
// ---------------------------------------------------------------------------
__global__ void pack_conv2_kernel(const bf16* __restrict__ x1, bf16* __restrict__ im)
{
  int i = blockIdx.x * blockDim.x + threadIdx.x;
  if (i >= 12000 * 3840) return;
  int row = i / 3840, col = i - row * 3840;
  int c = row / 100, t2 = row - c * 100;
  int ic = col / 3, kt = col - ic * 3;
  int u = 2 * t2 + kt - 1;
  bf16 val = (bf16)0.0f;
  if (u >= 0 && u < 200) val = x1[((size_t)c * 200 + u) * 1280 + ic];
  im[i] = val;
}

// ---------------------------------------------------------------------------
// h = conv2_out + sinusoidal positional embedding (per in-chunk position).
// ---------------------------------------------------------------------------
__global__ void add_posemb_kernel(const bf16* __restrict__ xin, bf16* __restrict__ hout)
{
  int i = blockIdx.x * blockDim.x + threadIdx.x;
  if (i >= 12000 * 1280) return;
  int row = i / 1280, d = i - row * 1280;
  int t = row % 100;
  const float inc = 9.2103403719761836f / 639.0f;   // log(10000)/(half-1)
  float pe;
  if (d < 640) pe = __sinf((float)t * __expf(-inc * (float)d));
  else         pe = __cosf((float)t * __expf(-inc * (float)(d - 640)));
  hout[i] = (bf16)((float)xin[i] + pe);
}

// ---------------------------------------------------------------------------
// AvgPool1d(2): (12000, 1280) -> (6000, 1280)
// ---------------------------------------------------------------------------
__global__ void avgpool2_kernel(const bf16* __restrict__ h, bf16* __restrict__ hp)
{
  int i = blockIdx.x * blockDim.x + threadIdx.x;
  if (i >= 6000 * 1280) return;
  int row = i / 1280, d = i - row * 1280;
  float a = (float)h[(size_t)(2 * row)     * 1280 + d];
  float b = (float)h[(size_t)(2 * row + 1) * 1280 + d];
  hp[i] = (bf16)(0.5f * (a + b));
}

// ---------------------------------------------------------------------------
extern "C" void kernel_launch(void* const* d_in, const int* in_sizes, int n_in,
                              void* d_out, int out_size, void* d_ws, size_t ws_size,
                              hipStream_t stream)
{
  const bf16*  mel     = (const bf16*) d_in[0];
  const bf16*  conv1_w = (const bf16*) d_in[1];
  const bf16*  conv1_b = (const bf16*) d_in[2];
  const bf16*  conv2_w = (const bf16*) d_in[3];
  const bf16*  conv2_b = (const bf16*) d_in[4];
  const float* ln1_w   = (const float*)d_in[5];
  const float* ln1_b   = (const float*)d_in[6];
  const bf16*  wq      = (const bf16*) d_in[7];
  const bf16*  bq      = (const bf16*) d_in[8];
  const bf16*  wk      = (const bf16*) d_in[9];
  const bf16*  wv      = (const bf16*) d_in[10];
  const bf16*  bv      = (const bf16*) d_in[11];
  const bf16*  wo      = (const bf16*) d_in[12];
  const bf16*  bo      = (const bf16*) d_in[13];
  const float* ln2_w   = (const float*)d_in[14];
  const float* ln2_b   = (const float*)d_in[15];
  const bf16*  w1      = (const bf16*) d_in[16];
  const bf16*  b1      = (const bf16*) d_in[17];
  const bf16*  w2      = (const bf16*) d_in[18];
  const bf16*  b2      = (const bf16*) d_in[19];
  const float* lnp_w   = (const float*)d_in[20];
  const float* lnp_b   = (const float*)d_in[21];
  const bf16*  proj_w  = (const bf16*) d_in[22];
  const bf16*  proj_b  = (const bf16*) d_in[23];
  (void)in_sizes; (void)n_in; (void)out_size; (void)ws_size;

  const int D = 1280, FF = 5120, OUTD = 3584, L = 32, M = 12000;
  char* ws = (char*)d_ws;
  const size_t SZ = (size_t)M * D * sizeof(bf16);     // 30,720,000 B
  bf16* h  = (bf16*)(ws + 0 * SZ);
  bf16* x  = (bf16*)(ws + 1 * SZ);
  bf16* qb = (bf16*)(ws + 2 * SZ);
  bf16* kb = (bf16*)(ws + 3 * SZ);
  bf16* vb = (bf16*)(ws + 4 * SZ);
  bf16* ob = (bf16*)(ws + 5 * SZ);
  bf16* tb = (bf16*)(ws + 6 * SZ);      // 12000*5120*2 = 122,880,000 B
  // conv-time aliases (dead regions at that point):
  bf16* c1im = qb;                      // 24000*384*2  fits in qb
  bf16* x1   = tb;                      // 24000*1280*2 fits in tb
  bf16* c2im = h;                       // 12000*3840*2 fits in h..qb span

  dim3 blk(256);
  auto gemm = [&](const bf16* A, const bf16* Wt, const bf16* bias, const bf16* res,
                  bf16* outp, int Mm, int Nn, int Kk, int act) {
    dim3 grid((Mm + 127) / 128, Nn / 128);
    gemm_bf16_kernel<<<grid, blk, 0, stream>>>(A, Wt, bias, res, outp, Mm, Nn, Kk, act);
  };

  // ---- conv front-end (im2col + WMMA GEMM + GELU) ----
  {
    int n1 = 24000 * 384;
    pack_conv1_kernel<<<dim3((n1 + 255) / 256), dim3(256), 0, stream>>>(mel, c1im);
    gemm(c1im, conv1_w, conv1_b, nullptr, x1, 24000, 1280, 384, 1);
    int n2 = 12000 * 3840;
    pack_conv2_kernel<<<dim3((n2 + 255) / 256), dim3(256), 0, stream>>>(x1, c2im);
    gemm(c2im, conv2_w, conv2_b, nullptr, ob, 12000, 1280, 3840, 1);
    int n3 = 12000 * 1280;
    add_posemb_kernel<<<dim3((n3 + 255) / 256), dim3(256), 0, stream>>>(ob, h);
  }

  // ---- 32 transformer layers ----
  for (int l = 0; l < L; ++l) {
    ln_kernel<<<dim3(M), dim3(128), 0, stream>>>(h, x, ln1_w + (size_t)l * D, ln1_b + (size_t)l * D, D);
    gemm(x, wq + (size_t)l * D * D, bq + (size_t)l * D, nullptr, qb, M, D, D, 0);
    gemm(x, wk + (size_t)l * D * D, nullptr,            nullptr, kb, M, D, D, 0);
    gemm(x, wv + (size_t)l * D * D, bv + (size_t)l * D, nullptr, vb, M, D, D, 0);
    attn_kernel<<<dim3(120, 20), dim3(128), 0, stream>>>(qb, kb, vb, ob);
    gemm(ob, wo + (size_t)l * D * D, bo + (size_t)l * D, h, h, M, D, D, 0);
    ln_kernel<<<dim3(M), dim3(128), 0, stream>>>(h, x, ln2_w + (size_t)l * D, ln2_b + (size_t)l * D, D);
    gemm(x,  w1 + (size_t)l * FF * D, b1 + (size_t)l * FF, nullptr, tb, M, FF, D, 1);
    gemm(tb, w2 + (size_t)l * D * FF, b2 + (size_t)l * D,  h,       h,  M, D, FF, 0);
  }

  // ---- tail: avgpool, post-LN, projection ----
  int n4 = 6000 * 1280;
  avgpool2_kernel<<<dim3((n4 + 255) / 256), dim3(256), 0, stream>>>(h, x);
  ln_kernel<<<dim3(6000), dim3(128), 0, stream>>>(x, qb, lnp_w, lnp_b, D);
  gemm(qb, proj_w, proj_b, nullptr, (bf16*)d_out, 6000, OUTD, D, 0);
}
